// MultiHeadAttention_18700287606866
// MI455X (gfx1250) — matrix operand order repaired and checked by reading, not compiled, not run on hardware
//
#include <hip/hip_runtime.h>
#include <hip/hip_bf16.h>

typedef __attribute__((ext_vector_type(16))) __bf16 v16bf;
typedef __attribute__((ext_vector_type(8)))  float  v8f;
typedef __attribute__((ext_vector_type(4)))  unsigned int v4u;
typedef __attribute__((ext_vector_type(8)))  int v8i;
typedef __attribute__((ext_vector_type(4)))  int v4i;

#define WMMA_BF16(a, b, c) \
  __builtin_amdgcn_wmma_f32_16x16x32_bf16(false, (a), false, (b), (short)0, (c), false, false)

static __device__ inline v8f vzero() {
  v8f z = {0.f, 0.f, 0.f, 0.f, 0.f, 0.f, 0.f, 0.f};
  return z;
}

// ---------------------------------------------------------------------------
// 16-lane butterfly reductions in pure VALU via DPP16 (no LDS traffic).
// quad_perm xor1 (0xB1), quad_perm xor2 (0x4E), row_half_mirror (0x141)
// acts as xor4 once quads are reduced, row_mirror (0x140) acts as xor8.
// ---------------------------------------------------------------------------
template <int CTRL>
static __device__ inline float dpp_mov(float x) {
  int r = __builtin_amdgcn_update_dpp(0, __builtin_bit_cast(int, x),
                                      CTRL, 0xf, 0xf, true);
  return __builtin_bit_cast(float, r);
}
static __device__ inline float red16_max(float x) {
  x = fmaxf(x, dpp_mov<0xB1>(x));
  x = fmaxf(x, dpp_mov<0x4E>(x));
  x = fmaxf(x, dpp_mov<0x141>(x));
  x = fmaxf(x, dpp_mov<0x140>(x));
  return x;
}
static __device__ inline float red16_sum(float x) {
  x += dpp_mov<0xB1>(x);
  x += dpp_mov<0x4E>(x);
  x += dpp_mov<0x141>(x);
  x += dpp_mov<0x140>(x);
  return x;
}

// ---------------------------------------------------------------------------
// TDM: 2-D bf16 tile load Global -> LDS (D# per CDNA5 ISA §8).
// ---------------------------------------------------------------------------
static __device__ inline void tdm_load_2d_bf16(unsigned lds_addr, const void* gptr,
                                               unsigned tile_d0, unsigned tile_d1,
                                               unsigned tensor_d0, unsigned tensor_d1,
                                               unsigned d0_stride,
                                               unsigned pad_interval_code,
                                               unsigned pad_amount_code) {
  unsigned long long ga = (unsigned long long)(size_t)gptr;
  v4u g0;
  g0[0] = 1u;                                            // count=1, no gather
  g0[1] = lds_addr;                                      // LDS byte address
  g0[2] = (unsigned)(ga & 0xffffffffu);                  // global_addr lo
  g0[3] = (unsigned)((ga >> 32) & 0x01ffffffu) | (2u << 30);  // addr hi | type=2
  v8i g1;
  unsigned w0 = (1u << 16)                               // data_size = 2 bytes
              | (1u << 20)                               // pad_enable
              | (pad_interval_code << 22)
              | (pad_amount_code << 25);
  g1[0] = (int)w0;
  g1[1] = (int)(tensor_d0 << 16);                        // tensor_dim0[15:0]
  g1[2] = (int)((tensor_d0 >> 16) | (tensor_d1 << 16));
  g1[3] = (int)((tensor_d1 >> 16) | (tile_d0 << 16));
  g1[4] = (int)(tile_d1 & 0xffffu);
  g1[5] = (int)d0_stride;
  g1[6] = 0;
  g1[7] = 0;
  v4i z4 = {0, 0, 0, 0};
  v8i z8 = {0, 0, 0, 0, 0, 0, 0, 0};
  __builtin_amdgcn_tensor_load_to_lds(g0, g1, z4, z4, z8, 0);
}

// ---------------------------------------------------------------------------
// GEMM: [8192,768] @ [768,768] + bias, bf16 WMMA, fp32 accumulate,
// double-buffered LDS (global prefetch overlaps WMMA).
// MODE 0: bf16 out [B,H,S,Hd] | MODE 2: bf16 out [B,H,Hd,S] | MODE 1: fp32 [M,N]
// ---------------------------------------------------------------------------
template <int MODE>
__global__ __launch_bounds__(256) void gemm_bias_bf16(const float* __restrict__ A,
                                                      const float* __restrict__ W,
                                                      const float* __restrict__ bias,
                                                      void* __restrict__ outv) {
  constexpr int K = 768, N = 768;
  constexpr int LDAs = 48;
  constexpr int LDBs = 48;
  __shared__ __bf16 lA[2][128 * LDAs];
  __shared__ __bf16 lB[2][128 * LDBs];

  const int tid   = threadIdx.x;
  const int mBase = blockIdx.y * 128;
  const int nBase = blockIdx.x * 128;
  const int lane  = tid & 31;
  const int wave  = tid >> 5;
  const int lh    = (lane >> 4) & 1;
  const int ln    = lane & 15;
  const int wm    = wave & 1;
  const int wn    = wave >> 1;

  v8f acc[4][2];
#pragma unroll
  for (int mt = 0; mt < 4; ++mt)
#pragma unroll
    for (int nt = 0; nt < 2; ++nt) acc[mt][nt] = vzero();

  float4 va[4], vw[4];
  // prologue: stage tile k=0 into buffer 0
#pragma unroll
  for (int i = 0; i < 4; ++i) {
    int lin = tid + i * 256;
    va[i] = *(const float4*)(A + (size_t)(mBase + (lin >> 3)) * K + (lin & 7) * 4);
    vw[i] = *(const float4*)(W + (size_t)(lin >> 5) * N + nBase + (lin & 31) * 4);
  }
#pragma unroll
  for (int i = 0; i < 4; ++i) {
    int lin = tid + i * 256;
    __bf16* d = &lA[0][(lin >> 3) * LDAs + (lin & 7) * 4];
    d[0] = (__bf16)va[i].x; d[1] = (__bf16)va[i].y;
    d[2] = (__bf16)va[i].z; d[3] = (__bf16)va[i].w;
    int k = lin >> 5, n4 = lin & 31;
    lB[0][(n4 * 4 + 0) * LDBs + k] = (__bf16)vw[i].x;
    lB[0][(n4 * 4 + 1) * LDBs + k] = (__bf16)vw[i].y;
    lB[0][(n4 * 4 + 2) * LDBs + k] = (__bf16)vw[i].z;
    lB[0][(n4 * 4 + 3) * LDBs + k] = (__bf16)vw[i].w;
  }

  int cur = 0;
  for (int kBase = 0; kBase < K; kBase += 32) {
    __syncthreads();  // buf[cur] ready; buf[cur^1] fully consumed

    // prefetch next K-slab into registers (wraps harmlessly on last iter)
    const int kn = (kBase + 32 < K) ? (kBase + 32) : 0;
#pragma unroll
    for (int i = 0; i < 4; ++i) {
      int lin = tid + i * 256;
      va[i] = *(const float4*)(A + (size_t)(mBase + (lin >> 3)) * K + kn + (lin & 7) * 4);
      vw[i] = *(const float4*)(W + (size_t)(kn + (lin >> 5)) * N + nBase + (lin & 31) * 4);
    }

    // fragments + WMMA from buf[cur]
    v16bf af[4];
#pragma unroll
    for (int mt = 0; mt < 4; ++mt) {
      const __bf16* p = &lA[cur][(wm * 64 + mt * 16 + ln) * LDAs];
#pragma unroll
      for (int i = 0; i < 16; ++i) af[mt][i] = p[lh * 16 + i];
    }
    v16bf bfg[2];
#pragma unroll
    for (int nt = 0; nt < 2; ++nt) {
      const __bf16* p = &lB[cur][(wn * 32 + nt * 16 + ln) * LDBs + lh * 16];
#pragma unroll
      for (int i = 0; i < 16; ++i) bfg[nt][i] = p[i];
    }
#pragma unroll
    for (int mt = 0; mt < 4; ++mt)
#pragma unroll
      for (int nt = 0; nt < 2; ++nt)
        acc[mt][nt] = WMMA_BF16(af[mt], bfg[nt], acc[mt][nt]);

    // store prefetched slab into the other buffer
#pragma unroll
    for (int i = 0; i < 4; ++i) {
      int lin = tid + i * 256;
      __bf16* d = &lA[cur ^ 1][(lin >> 3) * LDAs + (lin & 7) * 4];
      d[0] = (__bf16)va[i].x; d[1] = (__bf16)va[i].y;
      d[2] = (__bf16)va[i].z; d[3] = (__bf16)va[i].w;
      int k = lin >> 5, n4 = lin & 31;
      lB[cur ^ 1][(n4 * 4 + 0) * LDBs + k] = (__bf16)vw[i].x;
      lB[cur ^ 1][(n4 * 4 + 1) * LDBs + k] = (__bf16)vw[i].y;
      lB[cur ^ 1][(n4 * 4 + 2) * LDBs + k] = (__bf16)vw[i].z;
      lB[cur ^ 1][(n4 * 4 + 3) * LDBs + k] = (__bf16)vw[i].w;
    }
    cur ^= 1;
  }

#pragma unroll
  for (int nt = 0; nt < 2; ++nt) {
    int n = nBase + wn * 32 + nt * 16 + ln;
    float bv = bias[n];
#pragma unroll
    for (int mt = 0; mt < 4; ++mt) {
#pragma unroll
      for (int r = 0; r < 8; ++r) {
        int m = mBase + wm * 64 + mt * 16 + lh * 8 + r;
        float val = acc[mt][nt][r] + bv;
        if constexpr (MODE == 0) {
          int h = n >> 6, hd = n & 63;
          int bb = m >> 10, s = m & 1023;
          ((__bf16*)outv)[(((size_t)bb * 12 + h) * 1024 + s) * 64 + hd] = (__bf16)val;
        } else if constexpr (MODE == 2) {
          int h = n >> 6, hd = n & 63;
          int bb = m >> 10, s = m & 1023;
          ((__bf16*)outv)[(((size_t)bb * 12 + h) * 64 + hd) * 1024 + s] = (__bf16)val;
        } else {
          ((float*)outv)[(size_t)m * 768 + n] = val;
        }
      }
    }
  }
}

// ---------------------------------------------------------------------------
// Flash attention: grid (16, 96), 128 threads (4 waves), 16 query rows/wave.
// Double-buffered TDM: tile k+1 streams into LDS while waves WMMA on tile k.
// ---------------------------------------------------------------------------
__global__ __launch_bounds__(128) void flash_attn(const __bf16* __restrict__ Q,
                                                  const __bf16* __restrict__ Kb,
                                                  const __bf16* __restrict__ Vtb,
                                                  float* __restrict__ O) {
  constexpr int LDK = 72;  // [32 keys][64 hd], 144B rows (32 DW + 4 DW pad)
  constexpr int LDV = 48;  // [64 hd][32 keys],  96B rows (16 DW + 8 DW pad)
  constexpr int LDP = 48;  // per-wave P tile [16][32]
  __shared__ __bf16 lK[2][32 * LDK];
  __shared__ __bf16 lVT[2][64 * LDV];
  __shared__ __bf16 lP[4][16 * LDP];

  const int tid  = threadIdx.x;
  const int wave = tid >> 5, lane = tid & 31;
  const int lh   = (lane >> 4) & 1, ln = lane & 15;
  const int bh   = blockIdx.y;
  const int qTile = blockIdx.x * 64;
  const int qRow  = qTile + wave * 16;

  v16bf qf[2];
  {
    const __bf16* qp = Q + ((size_t)bh * 1024 + (qRow + ln)) * 64;
#pragma unroll
    for (int kk = 0; kk < 2; ++kk)
#pragma unroll
      for (int i = 0; i < 16; ++i) qf[kk][i] = qp[kk * 32 + lh * 16 + i];
  }

  float mrun[8], lrun[8];
#pragma unroll
  for (int r = 0; r < 8; ++r) { mrun[r] = -1e30f; lrun[r] = 0.f; }
  v8f oacc[4];
#pragma unroll
  for (int t = 0; t < 4; ++t) oacc[t] = vzero();

  const __bf16* kTileBase = Kb + (size_t)bh * 1024 * 64;
  const __bf16* vTileBase = Vtb + (size_t)bh * 64 * 1024;
  const unsigned lKa[2]  = {(unsigned)(size_t)(void*)lK[0],  (unsigned)(size_t)(void*)lK[1]};
  const unsigned lVTa[2] = {(unsigned)(size_t)(void*)lVT[0], (unsigned)(size_t)(void*)lVT[1]};

  // prologue: TDM tile 0 into buffer 0
  if (wave == 0) {
    tdm_load_2d_bf16(lKa[0], kTileBase, 64, 32, 64, 1024, 64, 4, 3);
    tdm_load_2d_bf16(lVTa[0], vTileBase, 32, 64, 1024, 64, 1024, 3, 7);
  }

  int cur = 0;
  for (int kt = 0; kt < 1024; kt += 32) {
    __syncthreads();  // all waves done reading buf[cur^1]
    if (wave == 0) __builtin_amdgcn_s_wait_tensorcnt(0);  // buf[cur] landed
    __syncthreads();  // publish buf[cur]
    if (wave == 0 && kt + 32 < 1024) {
      // stream next tile into the other buffer while we compute
      tdm_load_2d_bf16(lKa[cur ^ 1], kTileBase + (size_t)(kt + 32) * 64,
                       64, 32, 64, 1024, 64, 4, 3);
      tdm_load_2d_bf16(lVTa[cur ^ 1], vTileBase + (kt + 32),
                       32, 64, 1024, 64, 1024, 3, 7);
    }

    // ---- scores: load all 4 K-fragments, then 2 independent WMMA chains ----
    v16bf bk[2][2];
#pragma unroll
    for (int nt = 0; nt < 2; ++nt)
#pragma unroll
      for (int kk = 0; kk < 2; ++kk) {
        const __bf16* p = &lK[cur][(nt * 16 + ln) * LDK + kk * 32 + lh * 16];
#pragma unroll
        for (int i = 0; i < 16; ++i) bk[nt][kk][i] = p[i];
      }
    v8f s0 = WMMA_BF16(qf[0], bk[0][0], vzero());
    v8f s1 = WMMA_BF16(qf[0], bk[1][0], vzero());
    s0 = WMMA_BF16(qf[1], bk[0][1], s0);
    s1 = WMMA_BF16(qf[1], bk[1][1], s1);

    // ---- online softmax (scale 1/sqrt(Hd) folded in; DPP reductions) ----
#pragma unroll
    for (int r = 0; r < 8; ++r) {
      float t0 = s0[r] * 0.125f, t1 = s1[r] * 0.125f;
      float mx = red16_max(fmaxf(t0, t1));
      float mnew = fmaxf(mrun[r], mx);
      float p0 = __expf(t0 - mnew), p1 = __expf(t1 - mnew);
      float rs = red16_sum(p0 + p1);
      float esc = __expf(mrun[r] - mnew);
      lrun[r] = lrun[r] * esc + rs;
      mrun[r] = mnew;
#pragma unroll
      for (int t = 0; t < 4; ++t) oacc[t][r] *= esc;
      __bf16* pw = &lP[wave][(lh * 8 + r) * LDP];
      pw[ln]      = (__bf16)p0;
      pw[16 + ln] = (__bf16)p1;
    }

    asm volatile("s_wait_dscnt 0" ::: "memory");  // wave-private P transpose

    // ---- O += P @ V : load all fragments, 4 independent WMMA chains ----
    v16bf ap;
    {
      const __bf16* p = &lP[wave][ln * LDP];
#pragma unroll
      for (int i = 0; i < 16; ++i) ap[i] = p[lh * 16 + i];
    }
    v16bf bv[4];
#pragma unroll
    for (int t = 0; t < 4; ++t) {
      const __bf16* p = &lVT[cur][(t * 16 + ln) * LDV + lh * 16];
#pragma unroll
      for (int i = 0; i < 16; ++i) bv[t][i] = p[i];
    }
#pragma unroll
    for (int t = 0; t < 4; ++t) oacc[t] = WMMA_BF16(ap, bv[t], oacc[t]);

    cur ^= 1;
  }

  const int b = bh / 12, h = bh % 12;
#pragma unroll
  for (int t = 0; t < 4; ++t) {
    int hd = t * 16 + ln;
#pragma unroll
    for (int r = 0; r < 8; ++r) {
      int s = qTile + wave * 16 + lh * 8 + r;
      O[(((size_t)b * 1024 + s) * 12 + h) * 64 + hd] = oacc[t][r] / lrun[r];
    }
  }
}

// ---------------------------------------------------------------------------
extern "C" void kernel_launch(void* const* d_in, const int* in_sizes, int n_in,
                              void* d_out, int out_size, void* d_ws, size_t ws_size,
                              hipStream_t stream) {
  const float* x  = (const float*)d_in[0];
  const float* wq = (const float*)d_in[1];
  const float* bq = (const float*)d_in[2];
  const float* wk = (const float*)d_in[3];
  const float* bk = (const float*)d_in[4];
  const float* wv = (const float*)d_in[5];
  const float* bv = (const float*)d_in[6];
  const float* wo = (const float*)d_in[7];
  const float* bo = (const float*)d_in[8];

  const size_t per = (size_t)8192 * 768;
  char* wsb = (char*)d_ws;
  __bf16* qw  = (__bf16*)wsb;
  __bf16* kw  = qw + per;
  __bf16* vtw = kw + per;                             // [B,H,Hd,S]
  float*  aw  = (float*)(wsb + 3 * per * sizeof(__bf16));

  dim3 gg(768 / 128, 8192 / 128);  // (6, 64)
  gemm_bias_bf16<0><<<gg, 256, 0, stream>>>(x, wq, bq, qw);
  gemm_bias_bf16<0><<<gg, 256, 0, stream>>>(x, wk, bk, kw);
  gemm_bias_bf16<2><<<gg, 256, 0, stream>>>(x, wv, bv, vtw);
  flash_attn<<<dim3(16, 96), 128, 0, stream>>>(qw, kw, vtw, aw);
  gemm_bias_bf16<1><<<gg, 256, 0, stream>>>(aw, wo, bo, (float*)d_out);
}
